// ResidualVectorQuantizer_2834678415552
// MI455X (gfx1250) — matrix operand order repaired and checked by reading, not compiled, not run on hardware
//
#include <hip/hip_runtime.h>

#define B_ 32
#define D_ 256
#define T_ 2048
#define L_ 8
#define K_ 1024
#define N_ (B_ * T_)       // 65536 residual vectors
#define M_ 64              // rows per block (4 row-tiles of 16)
#define RT_ 4
#define RPAD_ 258          // padded f32 row stride (bank-spread)
#define BPAD_ 132          // padded u32 (bf16-pair) row stride (bank-spread)

typedef __attribute__((ext_vector_type(16))) __bf16        v16bf;
typedef __attribute__((ext_vector_type(8)))  float         v8f;
typedef __attribute__((ext_vector_type(8)))  unsigned int  v8u;

union BfPack { v8u u; v16bf b; };

__device__ __forceinline__ unsigned int f32_to_bf16(float f) {
  unsigned int u = __float_as_uint(f);
  return (u + 0x7FFFu + ((u >> 16) & 1u)) >> 16;   // round-to-nearest-even
}

// Order-preserving map f32 -> u32 (smaller float => smaller uint), branch-free.
__device__ __forceinline__ unsigned int f32_ord(float f) {
  unsigned int u = __float_as_uint(f);
  unsigned int m = (unsigned int)((int)u >> 31);   // all-ones if negative
  return u ^ (m | 0x80000000u);
}

// ---------------------------------------------------------------------------
// Prep: codebooks f32 -> packed bf16 in ws, plus |W|^2 per (layer, code).
// ---------------------------------------------------------------------------
__global__ void __launch_bounds__(256)
rvq_prep(const float* __restrict__ cb, unsigned int* __restrict__ cb16,
         float* __restrict__ wnorm) {
  const int wave = threadIdx.x >> 5;
  const int lane = threadIdx.x & 31;
  const int row  = blockIdx.x * 8 + wave;          // 0 .. L*K-1
  const float* src = cb + (size_t)row * D_ + lane * 8;

  float s = 0.0f;
  unsigned int pk[4];
#pragma unroll
  for (int i = 0; i < 4; ++i) {
    float f0 = src[2 * i], f1 = src[2 * i + 1];
    s += f0 * f0 + f1 * f1;
    pk[i] = f32_to_bf16(f0) | (f32_to_bf16(f1) << 16);
  }
  uint4 v = make_uint4(pk[0], pk[1], pk[2], pk[3]);
  *(uint4*)(cb16 + (size_t)row * (D_ / 2) + lane * 4) = v;

#pragma unroll
  for (int m = 16; m >= 1; m >>= 1) s += __shfl_xor(s, m, 32);
  if (lane == 0) wnorm[row] = s;
}

// ---------------------------------------------------------------------------
// Main: one block per 64-row tile; 8 waves, wave w owns 128 codes.
// bf16 mirror maintained incrementally (seeded at load, refreshed by the
// subtract step via 16-bit LDS stores). Per layer: B frags in regs reused
// over 4 LDS-loaded A row-tiles of WMMA -> u32 packed-key argmin
// (one butterfly per layer) -> f32 subtract. out = x - final residual.
// ---------------------------------------------------------------------------
__global__ void __launch_bounds__(256)
rvq_main(const float* __restrict__ x, const float* __restrict__ cb,
         const unsigned int* __restrict__ cb16, const float* __restrict__ wnorm,
         float* __restrict__ out, int* __restrict__ idx_out) {
  __shared__ float        lds_r[M_ * RPAD_];        // ~66 KB padded f32 residual
  __shared__ unsigned int lds_rbf[M_ * BPAD_];      // ~34 KB padded bf16 mirror
  __shared__ unsigned int lds_bk[8 * M_];           // per-wave best keys
  __shared__ int          lds_idx[M_];

  unsigned short* rbf16 = (unsigned short*)lds_rbf; // row stride 2*BPAD_ ushorts

  const int tid  = threadIdx.x;
  const int wave = tid >> 5;
  const int lane = tid & 31;
  const int hh   = lane >> 4;
  const int l15  = lane & 15;

  const int n0 = blockIdx.x * M_;    // 64 rows, same b (T % 64 == 0)
  const int b  = n0 / T_;
  const int t0 = n0 % T_;

  // Load x tile: r0[row][d] = x[b][d][t0+row]; seed f32 + bf16 mirror.
#pragma unroll 4
  for (int i = tid; i < M_ * D_; i += 256) {
    const int rr = i & (M_ - 1);
    const int d  = i >> 6;
    const float v = x[(size_t)b * D_ * T_ + (size_t)d * T_ + t0 + rr];
    lds_r[rr * RPAD_ + d] = v;
    rbf16[rr * (2 * BPAD_) + d] = (unsigned short)f32_to_bf16(v);
  }
  __syncthreads();

#pragma unroll 1
  for (int layer = 0; layer < L_; ++layer) {
    unsigned int bestk[RT_][8];
#pragma unroll
    for (int rt = 0; rt < RT_; ++rt)
#pragma unroll
      for (int i = 0; i < 8; ++i) bestk[rt][i] = 0xFFFFFFFFu;

#pragma unroll 1
    for (int sub = 0; sub < 8; ++sub) {
      const int col = wave * 128 + sub * 16 + l15;   // code index = B column
      const unsigned int* wrow = cb16 + ((size_t)layer * K_ + col) * (D_ / 2);

      // B fragments for all 8 k-steps: reused by 4 row-tiles.
      BfPack bm[8];
#pragma unroll
      for (int kt = 0; kt < 8; ++kt)
#pragma unroll
        for (int p = 0; p < 8; ++p)
          bm[kt].u[p] = wrow[kt * 16 + p + hh * 8];   // K = kt*32 + 2p + 16*half

      const float wn = wnorm[layer * K_ + col];

#pragma unroll
      for (int rt = 0; rt < RT_; ++rt) {
        const int arow = rt * 16 + l15;
        v8f acc = {};
#pragma unroll
        for (int kt = 0; kt < 8; ++kt) {
          BfPack a;
#pragma unroll
          for (int p = 0; p < 8; ++p) {
            // A 16x32 layout: pair index = kt*16 + p + 8*half
            a.u[p] = lds_rbf[arow * BPAD_ + kt * 16 + p + hh * 8];
          }
          acc = __builtin_amdgcn_wmma_f32_16x16x32_bf16(
              false, a.b, false, bm[kt].b, (short)0, acc, false, false);
        }
        // dist ~ |W|^2 - 2 r.W ; pack truncated ordered value + 10-bit col.
#pragma unroll
        for (int i = 0; i < 8; ++i) {
          const float v = fmaf(-2.0f, acc[i], wn);
          const unsigned int key = (f32_ord(v) & ~1023u) | (unsigned int)col;
          bestk[rt][i] = key < bestk[rt][i] ? key : bestk[rt][i];
        }
      }
    }

    // One butterfly per layer: min across the 16 columns (lanes) of each half.
#pragma unroll
    for (int rt = 0; rt < RT_; ++rt)
#pragma unroll
      for (int i = 0; i < 8; ++i) {
#pragma unroll
        for (int m = 1; m < 16; m <<= 1) {
          const unsigned int o = __shfl_xor(bestk[rt][i], m, 32);
          bestk[rt][i] = o < bestk[rt][i] ? o : bestk[rt][i];
        }
      }

    // Lane 0 of each half-wave publishes its rows' bests.
    if (l15 == 0) {
#pragma unroll
      for (int rt = 0; rt < RT_; ++rt)
#pragma unroll
        for (int i = 0; i < 8; ++i)
          lds_bk[wave * M_ + rt * 16 + hh * 8 + i] = bestk[rt][i];
    }
    __syncthreads();

    // Cross-wave argmin + index emit.
    if (tid < M_) {
      unsigned int bk = lds_bk[tid];
#pragma unroll
      for (int w = 1; w < 8; ++w) {
        const unsigned int k = lds_bk[w * M_ + tid];
        bk = k < bk ? k : bk;
      }
      const int bi = (int)(bk & 1023u);
      lds_idx[tid] = bi;
      idx_out[(size_t)b * (L_ * T_) + (size_t)layer * T_ + t0 + tid] = bi;
    }
    __syncthreads();

    // residual -= chosen code (f32), refreshing the bf16 mirror in place.
    {
      const int d = tid;
#pragma unroll 4
      for (int r = 0; r < M_; ++r) {
        const int code = lds_idx[r];
        const float nr = lds_r[r * RPAD_ + d] - cb[((size_t)layer * K_ + code) * D_ + d];
        lds_r[r * RPAD_ + d] = nr;
        rbf16[r * (2 * BPAD_) + d] = (unsigned short)f32_to_bf16(nr);
      }
    }
    __syncthreads();
  }

  // out = x - final residual (sum of quantized layers)
#pragma unroll 4
  for (int i = tid; i < M_ * D_; i += 256) {
    const int rr = i & (M_ - 1);
    const int d  = i >> 6;
    const size_t g = (size_t)b * D_ * T_ + (size_t)d * T_ + t0 + rr;
    out[g] = x[g] - lds_r[rr * RPAD_ + d];
  }
}

extern "C" void kernel_launch(void* const* d_in, const int* in_sizes, int n_in,
                              void* d_out, int out_size, void* d_ws, size_t ws_size,
                              hipStream_t stream) {
  (void)in_sizes; (void)n_in; (void)out_size; (void)ws_size;

  const float* x  = (const float*)d_in[0];   // (B, D, T) f32
  const float* cb = (const float*)d_in[1];   // (L, K, D) f32

  // ws layout: [0, 4MB) bf16 codebooks (packed u32), then 32KB f32 norms.
  unsigned int* cb16  = (unsigned int*)d_ws;
  float*        wnorm = (float*)((char*)d_ws + (size_t)L_ * K_ * (D_ / 2) * sizeof(unsigned int));

  float* out     = (float*)d_out;                          // (B, D, T) f32
  int*   idx_out = (int*)d_out + (size_t)B_ * D_ * T_;     // (B, L, T) i32

  rvq_prep<<<(L_ * K_) / 8, 256, 0, stream>>>(cb, cb16, wnorm);
  rvq_main<<<N_ / M_, 256, 0, stream>>>(x, cb, cb16, wnorm, out, idx_out);
}
